// E83CircularTowerCell_68135361184127
// MI455X (gfx1250) — compile-verified, matrix operand order checked by reading, not run on hardware
//
#include <hip/hip_runtime.h>
#include <hip/hip_bf16.h>
#include <math.h>

// Problem constants (from the reference)
#define T_STEPS 2048
#define BATCH   16
#define DIMK    1024
#define NDIM    64
#define KTOW    3
#define PROJ    448               // 3*2*64 (kv) + 64 (q)
#define ROWS    (T_STEPS * BATCH) // 32768

typedef __attribute__((ext_vector_type(16))) __bf16 v16bf;
typedef __attribute__((ext_vector_type(8)))  float  v8f;

union FragBF16 { uint4 u[2]; v16bf v; };

// ---------------------------------------------------------------------------
// Kernel 1: projection GEMM  proj[row][p] = x_flat[row][:] . W[p][:]
//   W = concat(W_kv (384x1024), W_q (64x1024)) along p.
//   bf16 WMMA (16x16x32) with f32 accumulation.
// Tiles: BM=128 x BN=64, BK=64. 256 threads = 8 waves; wave w owns rows
// [w*16, w*16+16) of the tile and all 4 column sub-tiles. All 20 fragment
// ds_load_b128s are issued before the 8 WMMAs so LDS latency overlaps
// matrix issue instead of s_wait_dscnt 0 per WMMA.
// ---------------------------------------------------------------------------
#define BM 128
#define BN 64
#define BK 64

__global__ __launch_bounds__(256)
void proj_gemm_wmma(const float* __restrict__ x,
                    const float* __restrict__ Wkv,
                    const float* __restrict__ Wq,
                    float* __restrict__ proj)
{
    __shared__ __align__(16) __bf16 As[BM][BK];  // 16 KB
    __shared__ __align__(16) __bf16 Bs[BN][BK];  //  8 KB (row n = output feature)

    const int tid  = threadIdx.x;
    const int wave = tid >> 5;
    const int lane = tid & 31;
    const int hlf  = lane >> 4;   // which half of the wave
    const int l16  = lane & 15;

    const int rowBase = blockIdx.x * BM;
    const int colBase = blockIdx.y * BN;

    v8f acc[4] = {};

    for (int k0 = 0; k0 < DIMK; k0 += BK) {
        // --- load + convert A tile: 128x64 f32 -> bf16 (32 elems / thread) ---
        {
            const int r = tid >> 1;            // 0..127
            const int h = (tid & 1) * 32;      // 0 or 32
            const float* src = x + (size_t)(rowBase + r) * DIMK + k0 + h;
            #pragma unroll
            for (int i = 0; i < 32; i += 4) {
                float4 f = *(const float4*)(src + i);
                As[r][h + i + 0] = (__bf16)f.x;
                As[r][h + i + 1] = (__bf16)f.y;
                As[r][h + i + 2] = (__bf16)f.z;
                As[r][h + i + 3] = (__bf16)f.w;
            }
        }
        // --- load + convert B tile: 64 features x 64 depth (16 elems / thread) ---
        {
            const int n = tid >> 2;            // 0..63
            const int h = (tid & 3) * 16;      // 0,16,32,48
            const int p = colBase + n;
            const float* wrow = (p < KTOW * 2 * NDIM)
                                    ? (Wkv + (size_t)p * DIMK)
                                    : (Wq  + (size_t)(p - KTOW * 2 * NDIM) * DIMK);
            const float* src = wrow + k0 + h;
            #pragma unroll
            for (int i = 0; i < 16; i += 4) {
                float4 f = *(const float4*)(src + i);
                Bs[n][h + i + 0] = (__bf16)f.x;
                Bs[n][h + i + 1] = (__bf16)f.y;
                Bs[n][h + i + 2] = (__bf16)f.z;
                Bs[n][h + i + 3] = (__bf16)f.w;
            }
        }
        __syncthreads();

        // Preload ALL fragments for this K-tile, then run 8 WMMAs back-to-back.
        // 16-bit A layout per 32-deep chunk: lane half 0 holds K = 0..7 then
        // 16..23; half 1 holds K = 8..15 then 24..31 => two contiguous
        // 8-element (16B) runs per chunk.
        FragBF16 a[2];
        {
            const __bf16* arow = &As[wave * 16 + l16][0];
            #pragma unroll
            for (int ck = 0; ck < 2; ++ck) {
                a[ck].u[0] = *(const uint4*)(arow + ck * 32 + 8 * hlf);
                a[ck].u[1] = *(const uint4*)(arow + ck * 32 + 16 + 8 * hlf);
            }
        }
        FragBF16 b[2][4];
        #pragma unroll
        for (int c = 0; c < 4; ++c) {
            const __bf16* brow = &Bs[c * 16 + l16][0];
            #pragma unroll
            for (int ck = 0; ck < 2; ++ck) {
                b[ck][c].u[0] = *(const uint4*)(brow + ck * 32 + 8 * hlf);
                b[ck][c].u[1] = *(const uint4*)(brow + ck * 32 + 16 + 8 * hlf);
            }
        }
        #pragma unroll
        for (int ck = 0; ck < 2; ++ck) {
            #pragma unroll
            for (int c = 0; c < 4; ++c) {
                acc[c] = __builtin_amdgcn_wmma_f32_16x16x32_bf16(
                    false, a[ck].v, false, b[ck][c].v, (short)0, acc[c],
                    false, false);
            }
        }
        __syncthreads();
    }

    // D layout: VGPR i -> row (hlf*8 + i), col = l16 within each 16x16 tile.
    #pragma unroll
    for (int c = 0; c < 4; ++c) {
        const int col = colBase + c * 16 + l16;
        #pragma unroll
        for (int i = 0; i < 8; ++i) {
            const int row = rowBase + wave * 16 + hlf * 8 + i;
            proj[(size_t)row * PROJ + col] = acc[c][i];
        }
    }
}

// ---------------------------------------------------------------------------
// Kernel 2: sequential circular-tower scan. One block per batch element b.
// 3x 64x64 f32 state matrices live in LDS (48 KB) for the whole scan.
// ---------------------------------------------------------------------------
__global__ __launch_bounds__(256)
void tower_scan(const float* __restrict__ proj,
                const float* __restrict__ M_init,
                const float* __restrict__ B_gates,
                float* __restrict__ out,       // [T,B,64]
                float* __restrict__ M_final)   // [K,B,64,64]
{
    __shared__ float Ms[KTOW][NDIM][NDIM];     // 48 KB state
    __shared__ float kns[KTOW][NDIM];
    __shared__ float vvs[KTOW][NDIM];
    __shared__ float qvs[NDIM];
    __shared__ float gks[KTOW][NDIM];
    __shared__ float cgi[KTOW][NDIM];
    __shared__ float rts[KTOW][NDIM];
    __shared__ float rgs[KTOW][NDIM];
    __shared__ float cgs[KTOW][NDIM];
    __shared__ float dls[KTOW][NDIM];
    __shared__ float bgs[KTOW][NDIM];
    __shared__ float inv_norm[KTOW];

    const int tid = threadIdx.x;
    const int b   = blockIdx.x;

    // Load initial state and gate biases.
    for (int e = tid; e < KTOW * NDIM * NDIM; e += 256) {
        const int kk = e / (NDIM * NDIM);
        const int m  = e % (NDIM * NDIM);
        Ms[kk][m / NDIM][m % NDIM] =
            M_init[((size_t)(kk * BATCH + b) * NDIM + m / NDIM) * NDIM + m % NDIM];
    }
    for (int e = tid; e < KTOW * NDIM; e += 256)
        bgs[e / NDIM][e % NDIM] = B_gates[e];
    __syncthreads();

    for (int t = 0; t < T_STEPS; ++t) {
        const size_t base = ((size_t)t * BATCH + b) * PROJ;

        // Fetch k, v, q vectors for this step.
        if (tid < KTOW * NDIM) {
            const int kk = tid / NDIM, n = tid % NDIM;
            kns[kk][n] = proj[base + kk * 2 * NDIM + n];
            vvs[kk][n] = proj[base + kk * 2 * NDIM + NDIM + n];
        } else if (tid < KTOW * NDIM + NDIM) {
            const int n = tid - KTOW * NDIM;
            qvs[n] = proj[base + KTOW * 2 * NDIM + n];
        }
        __syncthreads();

        // Norms of the 3 key vectors.
        if (tid < KTOW) {
            float s = 0.f;
            #pragma unroll 8
            for (int j = 0; j < NDIM; ++j) { const float v = kns[tid][j]; s += v * v; }
            inv_norm[tid] = 1.f / (sqrtf(s) + 1e-6f);
        }
        __syncthreads();
        if (tid < KTOW * NDIM) kns[tid / NDIM][tid % NDIM] *= inv_norm[tid / NDIM];
        __syncthreads();

        // Matvecs on the OLD state: gater = M[(k+1)%3].
        if (tid < KTOW * NDIM) {
            const int kk = tid / NDIM, i = tid % NDIM;
            const int kg = (kk + 1) % KTOW;
            float sg = 0.f, sc = 0.f, sr = 0.f;
            #pragma unroll 4
            for (int j = 0; j < NDIM; ++j) {
                const float kv = kns[kk][j];
                sg += Ms[kg][i][j] * kv;   // row gate input
                sc += Ms[kg][j][i] * kv;   // col gate input
                sr += Ms[kk][i][j] * kv;   // retrieved
            }
            gks[kk][i] = sg; cgi[kk][i] = sc; rts[kk][i] = sr;
        }
        __syncthreads();

        // Gates and delta.
        if (tid < KTOW * NDIM) {
            const int kk = tid / NDIM, i = tid % NDIM;
            rgs[kk][i] = 1.f / (1.f + __expf(-(gks[kk][i] + bgs[kk][i])));
            cgs[kk][i] = 1.f / (1.f + __expf(-(cgi[kk][i] + bgs[kk][i])));
            dls[kk][i] = vvs[kk][i] - rts[kk][i];
        }
        __syncthreads();

        // State update: M = rg_i * M * cg_j + delta_i * kn_j.
        for (int e = tid; e < KTOW * NDIM * NDIM; e += 256) {
            const int kk = e / (NDIM * NDIM);
            const int m  = e % (NDIM * NDIM);
            const int i = m / NDIM, j = m % NDIM;
            Ms[kk][i][j] = rgs[kk][i] * Ms[kk][i][j] * cgs[kk][j]
                         + dls[kk][i] * kns[kk][j];
        }
        __syncthreads();

        // Readout from tower 0: out = Sq * silu(Sq) = Sq^2 * sigmoid(Sq).
        if (tid < NDIM) {
            float s = 0.f;
            #pragma unroll 8
            for (int j = 0; j < NDIM; ++j) s += Ms[0][tid][j] * qvs[j];
            const float sig = 1.f / (1.f + __expf(-s));
            out[((size_t)t * BATCH + b) * NDIM + tid] = s * s * sig;
        }
        __syncthreads();
    }

    // Write final state.
    for (int e = tid; e < KTOW * NDIM * NDIM; e += 256) {
        const int kk = e / (NDIM * NDIM);
        const int m  = e % (NDIM * NDIM);
        M_final[((size_t)(kk * BATCH + b) * NDIM + m / NDIM) * NDIM + m % NDIM] =
            Ms[kk][m / NDIM][m % NDIM];
    }
}

// ---------------------------------------------------------------------------
extern "C" void kernel_launch(void* const* d_in, const int* in_sizes, int n_in,
                              void* d_out, int out_size, void* d_ws, size_t ws_size,
                              hipStream_t stream) {
    const float* x      = (const float*)d_in[0];  // (T,B,DIM)
    const float* M_init = (const float*)d_in[1];  // (K,B,N,N)
    const float* W_kv   = (const float*)d_in[2];  // (K*2*N, DIM)
    const float* W_q    = (const float*)d_in[3];  // (N, DIM)
    const float* B_g    = (const float*)d_in[4];  // (K,N)

    float* out     = (float*)d_out;                                   // (T,B,N)
    float* M_final = out + (size_t)T_STEPS * BATCH * NDIM;            // (K,B,N,N)
    float* proj    = (float*)d_ws;                                    // ROWS x 448 f32

    dim3 g1(ROWS / BM, PROJ / BN);   // 256 x 7 blocks
    proj_gemm_wmma<<<g1, 256, 0, stream>>>(x, W_kv, W_q, proj);
    tower_scan<<<BATCH, 256, 0, stream>>>(proj, M_init, B_g, out, M_final);
}
